// HGRN2Attention_32839319945674
// MI455X (gfx1250) — matrix operand order repaired and checked by reading, not compiled, not run on hardware
//
#include <hip/hip_runtime.h>
#include <hip/hip_bf16.h>
#include <stdint.h>

// ---------------------------------------------------------------------------
// HGRN2 attention block for MI455X (gfx1250), wave32 + WMMA bf16.
// ---------------------------------------------------------------------------

typedef __attribute__((ext_vector_type(16))) __bf16 v16bf;
typedef __attribute__((ext_vector_type(8)))  float  v8f;

constexpr int Bb   = 4;
constexpr int Ss   = 2048;
constexpr int Dd   = 1024;
constexpr int Hh   = 16;
constexpr int HDh  = 64;
constexpr int Mtok = Bb * Ss;      // 8192 tokens
constexpr int Ncat = 5 * Dd;       // 5120 fused projection columns
constexpr float EPSv = 1e-5f;

// ---------------------------------------------------------------------------
// Kernel 1: fp32 -> bf16 convert (grid stride)
// ---------------------------------------------------------------------------
__global__ __launch_bounds__(256)
void k_f32_to_bf16(const float* __restrict__ src, __hip_bfloat16* __restrict__ dst, int n) {
    for (int i = blockIdx.x * blockDim.x + threadIdx.x; i < n; i += gridDim.x * blockDim.x)
        dst[i] = __float2bfloat16(src[i]);
}

// ---------------------------------------------------------------------------
// Kernel 2: transpose W [K,N] (row-major, as used by x@W) into Wt [N,K] bf16.
// Wt row n is column n of W -> contiguous-K rows, the layout WMMA B wants.
// ---------------------------------------------------------------------------
__global__ __launch_bounds__(256)
void k_pack_wt(const float* __restrict__ W, __hip_bfloat16* __restrict__ Wt, int K, int N) {
    int total = K * N;
    for (int i = blockIdx.x * blockDim.x + threadIdx.x; i < total; i += gridDim.x * blockDim.x) {
        int n = i / K;
        int k = i - n * K;
        Wt[(size_t)n * K + k] = __float2bfloat16(W[(size_t)k * N + n]);
    }
}

// ---------------------------------------------------------------------------
// Kernel 3/6: WMMA bf16 GEMM.  C[M,N] = A[M,K] * Bt[N,K]^T
// Per wave: 32x64 output tile, K-step 32 -> 8 x v_wmma_f32_16x16x32_bf16.
// mode==1: projection epilogue (sigmoid(+bias) on cols [3072,4096),
//          silu on cols >= 4096).
// Fragment layouts per CDNA5 ISA 7.12.2:
//  A (16x32 bf16): lane<16 row M=lane, elems 0..7 = K0..7, 8..15 = K16..23;
//                  lane>=16 same row, K8..15 and K24..31.
//  B (32x16 bf16) from Wt: lane<16 col N=lane holds K0..7, K16..23;
//                  lane>=16 holds K8..15, K24..31.
//  C (16x16 f32): VGPR r -> lane<16: (M=r, N=lane); lane>=16: (M=r+8, N=lane-16)
// ---------------------------------------------------------------------------
__global__ __launch_bounds__(256)
void k_gemm_wmma_bf16(const __hip_bfloat16* __restrict__ A,
                      const __hip_bfloat16* __restrict__ Bt,
                      float* __restrict__ C,
                      const float* __restrict__ fbias,
                      int M, int N, int K, int mode) {
    const int wave   = blockIdx.x * (blockDim.x >> 5) + (threadIdx.x >> 5);
    const int lid    = threadIdx.x & 31;
    const int ntiles = N >> 6;           // 64-wide N tiles
    const int mtiles = M >> 5;           // 32-tall M tiles
    if (wave >= mtiles * ntiles) return;

    const int mt    = wave / ntiles;
    const int nt    = wave - mt * ntiles;
    const int mBase = mt << 5;
    const int nBase = nt << 6;

    const int  lmod = lid & 15;
    const bool hi   = lid >= 16;

    const v8f zero = {0.f, 0.f, 0.f, 0.f, 0.f, 0.f, 0.f, 0.f};
    v8f acc[2][4];
    #pragma unroll
    for (int i = 0; i < 2; ++i)
        #pragma unroll
        for (int j = 0; j < 4; ++j) acc[i][j] = zero;

    const size_t rowA0 = (size_t)(mBase + lmod) * K;
    const size_t rowA1 = rowA0 + (size_t)16 * K;
    const int    aOff  = hi ? 8  : 0;
    const int    bOff  = hi ? 8  : 0;

    size_t rowB[4];
    #pragma unroll
    for (int j = 0; j < 4; ++j) rowB[j] = (size_t)(nBase + j * 16 + lmod) * K;

    for (int k0 = 0; k0 < K; k0 += 32) {
        union Frag { v16bf bf; uint4 q[2]; };
        Frag a0, a1, bfr[4];

        const __hip_bfloat16* pa0 = A + rowA0 + k0 + aOff;
        const __hip_bfloat16* pa1 = A + rowA1 + k0 + aOff;
        a0.q[0] = *reinterpret_cast<const uint4*>(pa0);
        a0.q[1] = *reinterpret_cast<const uint4*>(pa0 + 16);
        a1.q[0] = *reinterpret_cast<const uint4*>(pa1);
        a1.q[1] = *reinterpret_cast<const uint4*>(pa1 + 16);

        #pragma unroll
        for (int j = 0; j < 4; ++j) {
            const __hip_bfloat16* pb = Bt + rowB[j] + k0 + bOff;
            bfr[j].q[0] = *reinterpret_cast<const uint4*>(pb);
            bfr[j].q[1] = *reinterpret_cast<const uint4*>(pb + 16);
        }

        // pull next K-slice of A toward the caches (global_prefetch_b8)
        __builtin_prefetch(pa0 + 32, 0, 1);
        __builtin_prefetch(pa1 + 32, 0, 1);

        #pragma unroll
        for (int j = 0; j < 4; ++j) {
            acc[0][j] = __builtin_amdgcn_wmma_f32_16x16x32_bf16(
                false, a0.bf, false, bfr[j].bf, (short)0, acc[0][j], false, false);
            acc[1][j] = __builtin_amdgcn_wmma_f32_16x16x32_bf16(
                false, a1.bf, false, bfr[j].bf, (short)0, acc[1][j], false, false);
        }
    }

    #pragma unroll
    for (int half = 0; half < 2; ++half) {
        const int mrow0 = mBase + half * 16 + (hi ? 8 : 0);
        #pragma unroll
        for (int j = 0; j < 4; ++j) {
            const int col = nBase + j * 16 + lmod;
            #pragma unroll
            for (int r = 0; r < 8; ++r) {
                float v = acc[half][j][r];
                if (mode == 1) {
                    if (col >= 4096) {                       // g-proj: silu
                        v = v / (1.f + __expf(-v));
                    } else if (col >= 3072) {                // f-proj: sigmoid(+b)
                        float t = v + fbias[col - 3072];
                        v = 1.f / (1.f + __expf(-t));
                    }
                }
                C[(size_t)(mrow0 + r) * N + col] = v;
            }
        }
    }
}

// ---------------------------------------------------------------------------
// Kernel 4: HGRN2 recurrent scan — wave-autonomous, barrier-free.
// One wave32 owns TWO state rows (i, i+1) of one (b,h):
//   lanes 0..15  : row i,   each lane holds 4 state columns (j = 4*lane .. +3)
//   lanes 16..31 : row i+1, same column slices
// Per step: st = st*g[i] + v[i]*k[j]; o[i] = sum_j st*q[j] via 4x shfl_xor
// (masks 1,2,4,8 stay inside each 16-lane half). Next step's q/k/v/g are
// register-prefetched so the load latency hides under the current step.
// 2048 independent waves (256 blocks) instead of 64 barrier-bound blocks.
// ---------------------------------------------------------------------------
__global__ __launch_bounds__(256)
void k_hgrn2_scan(const float* __restrict__ P, float* __restrict__ attn) {
    const int w    = blockIdx.x * (blockDim.x >> 5) + (threadIdx.x >> 5);
    const int lane = threadIdx.x & 31;
    const int bh   = w >> 5;                 // 32 waves per (b,h)
    const int rp   = w & 31;                 // row-pair index within the head
    const int b    = bh >> 4;
    const int h    = bh & 15;
    const int i0   = rp * 2 + (lane >> 4);   // this lane's state row
    const int jb   = (lane & 15) * 4;        // this lane's 4-wide j slice

    size_t base = ((size_t)b * Ss) * Ncat + (size_t)h * HDh;

    float st0 = 0.f, st1 = 0.f, st2 = 0.f, st3 = 0.f;

    // prologue: step 0 operands
    float4 q = *reinterpret_cast<const float4*>(P + base + jb);
    float4 k = *reinterpret_cast<const float4*>(P + base + 1024 + jb);
    float  v = P[base + 2048 + i0];
    float  g = P[base + 3072 + i0];

    for (int s = 0; s < Ss; ++s) {
        const size_t nbase = base + Ncat;
        float4 qn = {0.f, 0.f, 0.f, 0.f}, kn = {0.f, 0.f, 0.f, 0.f};
        float  vn = 0.f, gn = 0.f;
        if (s + 1 < Ss) {                    // issue next step's loads first
            qn = *reinterpret_cast<const float4*>(P + nbase + jb);
            kn = *reinterpret_cast<const float4*>(P + nbase + 1024 + jb);
            vn = P[nbase + 2048 + i0];
            gn = P[nbase + 3072 + i0];
        }

        st0 = st0 * g + v * k.x;  float acc = st0 * q.x;
        st1 = st1 * g + v * k.y;  acc += st1 * q.y;
        st2 = st2 * g + v * k.z;  acc += st2 * q.z;
        st3 = st3 * g + v * k.w;  acc += st3 * q.w;

        acc += __shfl_xor(acc, 1, 32);
        acc += __shfl_xor(acc, 2, 32);
        acc += __shfl_xor(acc, 4, 32);
        acc += __shfl_xor(acc, 8, 32);
        if ((lane & 15) == 0)
            attn[((size_t)(b * Ss + s)) * Dd + h * HDh + i0] = acc;

        q = qn; k = kn; v = vn; g = gn;
        base = nbase;
    }
}

// ---------------------------------------------------------------------------
// Kernel 5: LayerNorm over D=1024, * gamma + beta, * silu(x@Wg) (precomputed
// in P columns [4096,5120)), emit bf16 for the final WMMA GEMM.
// ---------------------------------------------------------------------------
__global__ __launch_bounds__(256)
void k_ln_gate(const float* __restrict__ attn, const float* __restrict__ P,
               const float* __restrict__ gamma, const float* __restrict__ beta,
               __hip_bfloat16* __restrict__ gated) {
    const int row = blockIdx.x;
    const int tid = threadIdx.x;

    float vals[4], s = 0.f, s2 = 0.f;
    #pragma unroll
    for (int i = 0; i < 4; ++i) {
        const int c = tid + i * 256;
        const float v = attn[(size_t)row * Dd + c];
        vals[i] = v; s += v; s2 += v * v;
    }
    __shared__ float rs_[256], rq_[256];
    rs_[tid] = s; rq_[tid] = s2;
    __syncthreads();
    for (int off = 128; off > 0; off >>= 1) {
        if (tid < off) { rs_[tid] += rs_[tid + off]; rq_[tid] += rq_[tid + off]; }
        __syncthreads();
    }
    const float mu   = rs_[0] * (1.f / Dd);
    const float var  = rq_[0] * (1.f / Dd) - mu * mu;
    const float rstd = rsqrtf(var + EPSv);

    #pragma unroll
    for (int i = 0; i < 4; ++i) {
        const int c = tid + i * 256;
        const float y = (vals[i] - mu) * rstd * gamma[c] + beta[c];
        const float gt = P[(size_t)row * Ncat + 4096 + c];   // silu already applied
        gated[(size_t)row * Dd + c] = __float2bfloat16(y * gt);
    }
}

// ---------------------------------------------------------------------------
// Launch
// ---------------------------------------------------------------------------
extern "C" void kernel_launch(void* const* d_in, const int* in_sizes, int n_in,
                              void* d_out, int out_size, void* d_ws, size_t ws_size,
                              hipStream_t stream) {
    const float* x     = (const float*)d_in[0];
    const float* Wq    = (const float*)d_in[1];
    const float* Wk    = (const float*)d_in[2];
    const float* Wv    = (const float*)d_in[3];
    const float* Wf    = (const float*)d_in[4];
    const float* bfv   = (const float*)d_in[5];
    const float* Wg    = (const float*)d_in[6];
    const float* Wout  = (const float*)d_in[7];
    const float* gamma = (const float*)d_in[8];
    const float* beta  = (const float*)d_in[9];
    float* out = (float*)d_out;

    // workspace layout
    char* ws = (char*)d_ws;
    constexpr size_t OFF_XB    = 0;
    constexpr size_t SZ_XB    = (size_t)Mtok * Dd * 2;
    constexpr size_t OFF_WCT  = OFF_XB  + SZ_XB;
    constexpr size_t SZ_WCT   = (size_t)Ncat * Dd * 2;
    constexpr size_t OFF_WOT  = OFF_WCT + SZ_WCT;
    constexpr size_t SZ_WOT   = (size_t)Dd * Dd * 2;
    constexpr size_t OFF_P    = OFF_WOT + SZ_WOT;
    constexpr size_t SZ_P     = (size_t)Mtok * Ncat * 4;
    constexpr size_t OFF_ATT  = OFF_P   + SZ_P;
    constexpr size_t SZ_ATT   = (size_t)Mtok * Dd * 4;
    constexpr size_t OFF_GATE = OFF_ATT + SZ_ATT;

    __hip_bfloat16* xb    = (__hip_bfloat16*)(ws + OFF_XB);
    __hip_bfloat16* WcatT = (__hip_bfloat16*)(ws + OFF_WCT);
    __hip_bfloat16* WoutT = (__hip_bfloat16*)(ws + OFF_WOT);
    float*          P     = (float*)(ws + OFF_P);
    float*          attn  = (float*)(ws + OFF_ATT);
    __hip_bfloat16* gated = (__hip_bfloat16*)(ws + OFF_GATE);

    // 1) x -> bf16
    k_f32_to_bf16<<<1024, 256, 0, stream>>>(x, xb, Mtok * Dd);

    // 2) pack W^T (bf16): [Wq|Wk|Wv|Wf|Wg] -> WcatT rows, Wout -> WoutT
    const float* Wlist[5] = {Wq, Wk, Wv, Wf, Wg};
    for (int w = 0; w < 5; ++w)
        k_pack_wt<<<512, 256, 0, stream>>>(Wlist[w], WcatT + (size_t)w * Dd * Dd, Dd, Dd);
    k_pack_wt<<<512, 256, 0, stream>>>(Wout, WoutT, Dd, Dd);

    // 3) fused projection GEMM: P[8192,5120] = x * Wcat (epilogue: sigmoid/silu)
    {
        const int waves  = (Mtok / 32) * (Ncat / 64);  // 20480
        const int blocks = waves / 8;                  // 2560
        k_gemm_wmma_bf16<<<blocks, 256, 0, stream>>>(xb, WcatT, P, bfv,
                                                     Mtok, Ncat, Dd, /*mode=*/1);
    }

    // 4) recurrent scan: 2048 wave-autonomous waves (2 rows each), 256 blocks
    {
        const int waves  = Bb * Hh * (HDh / 2);        // 2048
        const int blocks = waves / 8;                  // 256
        k_hgrn2_scan<<<blocks, 256, 0, stream>>>(P, attn);
    }

    // 5) LayerNorm + output gate -> bf16
    k_ln_gate<<<Mtok, 256, 0, stream>>>(attn, P, gamma, beta, gated);

    // 6) final GEMM: out[8192,1024] = gated * Wout
    {
        const int waves  = (Mtok / 32) * (Dd / 64);    // 4096
        const int blocks = waves / 8;                  // 512
        k_gemm_wmma_bf16<<<blocks, 256, 0, stream>>>(gated, WoutT, out, nullptr,
                                                     Mtok, Dd, Dd, /*mode=*/0);
    }
}
